// MoEFFN_60146722013334
// MI455X (gfx1250) — matrix operand order repaired and checked by reading, not compiled, not run on hardware
//
#include <hip/hip_runtime.h>
#include <math.h>

// ---------------------------------------------------------------------------
// MoE FFN (top-1) for gfx1250 / MI455X.
//   tokens=4096, D_MODEL=1024, D_HIDDEN=4096, E=8, fp32 io.
// route -> bucket tokens per expert -> grouped GEMMs with
// v_wmma_f32_16x16x32_bf16 (fp32 -> bf16 in flight, fp32 accumulation).
// Block tile: M=64 tokens x N=256, 8 waves, each wave 16x128 (8 accumulators).
// B fragments preloaded in groups of 4 (distinct regs) so ds_load_b128s are
// clause-batched and WMMAs issue back-to-back instead of waiting per-fragment.
// Double-buffered LDS (padded rows -> conflict-free), 1 barrier per k-step.
// ---------------------------------------------------------------------------

typedef __bf16 v16bf  __attribute__((ext_vector_type(16)));
typedef __bf16 bf16x8 __attribute__((ext_vector_type(8)));
typedef __bf16 bf16x4 __attribute__((ext_vector_type(4)));
typedef float  v8f    __attribute__((ext_vector_type(8)));

#define DM    1024            // d_model
#define DH    4096            // d_hidden
#define NE    8
#define NTOK  4096
#define TM    64              // tokens per block
#define TNB   256             // N per block
#define KS    32              // K per WMMA step
#define MAXT  64              // worst-case M tiles per expert (4096/64)
#define LDT   40              // padded LDS row stride (80B -> bank step 20)
#define ASZ   (TM * LDT)      // 2560 elems per A buffer
#define BSZ   (TNB * LDT)     // 10240 elems per B buffer

// ---------------------------------------------------------------- router ----
__global__ __launch_bounds__(256) void moe_router_k(
    const float* __restrict__ x, const float* __restrict__ Wg,
    const float* __restrict__ bg, int* __restrict__ cnt, int* __restrict__ bucket)
{
    const int t = blockIdx.x * 256 + threadIdx.x;
    if (t >= NTOK) return;
    float acc[NE];
#pragma unroll
    for (int e = 0; e < NE; ++e) acc[e] = bg[e];
    const float* xr = x + (size_t)t * DM;
    for (int d = 0; d < DM; d += 4) {
        const float4 xv = *(const float4*)(xr + d);
#pragma unroll
        for (int e = 0; e < NE; ++e) {
            acc[e] = fmaf(xv.x, Wg[(d + 0) * NE + e], acc[e]);
            acc[e] = fmaf(xv.y, Wg[(d + 1) * NE + e], acc[e]);
            acc[e] = fmaf(xv.z, Wg[(d + 2) * NE + e], acc[e]);
            acc[e] = fmaf(xv.w, Wg[(d + 3) * NE + e], acc[e]);
        }
    }
    int best = 0; float bv = acc[0];
#pragma unroll
    for (int e = 1; e < NE; ++e)              // strict '>' => first max (argmax)
        if (acc[e] > bv) { bv = acc[e]; best = e; }
    const int pos = atomicAdd(&cnt[best], 1);
    bucket[best * NTOK + pos] = t;
}

// ---------------------------------------------------------- wmma helpers ----
__device__ __forceinline__ v16bf frag16(const __bf16* p0, const __bf16* p1) {
    bf16x8 lo = *(const bf16x8*)p0;
    bf16x8 hi = *(const bf16x8*)p1;
    return __builtin_shufflevector(lo, hi, 0,1,2,3,4,5,6,7,8,9,10,11,12,13,14,15);
}

// ------------------------------------------------------------------ FFN1 ----
// h[token][n] = gelu(x[token] @ W1[e] + b1[e]) -> bf16. K=1024, N-blocks=16.
__global__ __launch_bounds__(256) void moe_ffn1_k(
    const float* __restrict__ x, const float* __restrict__ W1,
    const float* __restrict__ b1, const int* __restrict__ cnt,
    const int* __restrict__ bucket, __bf16* __restrict__ h)
{
    const int e    = blockIdx.x >> 6;
    const int tile = blockIdx.x & (MAXT - 1);
    const int count = cnt[e];
    if (tile * TM >= count) return;                       // uniform early-exit
    const int n0 = blockIdx.y * TNB;

    __shared__ __align__(16) __bf16 As[2 * ASZ];          // [buf][m][k], padded
    __shared__ __align__(16) __bf16 Bs[2 * BSZ];          // [buf][n][k], padded
    __shared__ int toks[TM];

    const int tid = threadIdx.x;
    if (tid < TM) {
        const int slot = tile * TM + tid;
        toks[tid] = (slot < count) ? bucket[e * NTOK + slot] : 0;
    }
    __syncthreads();

    // A staging: thread owns (m = tid>>2, 8 k's at ak = (tid&3)*8).
    const int am = tid >> 2, ak = (tid & 3) * 8;
    const float* xrow = x + (size_t)toks[am] * DM;
    // B staging: thread owns column n = n0 + tid; stages all 32 k's per step.
    const float* wcol = W1 + (size_t)e * DM * DH + n0 + tid;

    // Fragments: wave w -> mw = w&3 (M subtile), nwv = w>>2 (128-wide N strip)
    const int lane = tid & 31, wave = tid >> 5;
    const int half = lane >> 4, nm = lane & 15;
    const int mw = wave & 3, nwv = wave >> 2;
    const int aoff = (mw * 16 + nm) * LDT + half * 16;    // A: K {h*16..+15}
    const int boff = (nwv * 128 + nm) * LDT + half * 16;  // B: K {h*16..+15}

    v8f c[8] = {{}, {}, {}, {}, {}, {}, {}, {}};
    float4 av0, av1;
    float br0[16], br1[16];

    auto load_regs = [&](int k0) {
        av0 = *(const float4*)(xrow + k0 + ak);
        av1 = *(const float4*)(xrow + k0 + ak + 4);
        const float* wp = wcol + (size_t)k0 * DH;
#pragma unroll
        for (int p = 0; p < 16; ++p) {                    // rows k0+2p, k0+2p+1
            br0[p] = wp[(size_t)(2 * p) * DH];
            br1[p] = wp[(size_t)(2 * p + 1) * DH];
        }
    };
    auto store_regs = [&](int buf) {
        bf16x8 a8;
        a8[0] = (__bf16)av0.x; a8[1] = (__bf16)av0.y;
        a8[2] = (__bf16)av0.z; a8[3] = (__bf16)av0.w;
        a8[4] = (__bf16)av1.x; a8[5] = (__bf16)av1.y;
        a8[6] = (__bf16)av1.z; a8[7] = (__bf16)av1.w;
        *(bf16x8*)&As[buf * ASZ + am * LDT + ak] = a8;    // one ds_store_b128
#pragma unroll
        for (int p = 0; p < 16; p += 2) {                 // pairs merge -> b128
            bf16x4 b4;
            b4[0] = (__bf16)br0[p];     b4[1] = (__bf16)br1[p];
            b4[2] = (__bf16)br0[p + 1]; b4[3] = (__bf16)br1[p + 1];
            *(bf16x4*)&Bs[buf * BSZ + tid * LDT + 2 * p] = b4;
        }
    };
    auto compute = [&](int buf) {
        const __bf16* Ab = As + buf * ASZ + aoff;
        const __bf16* Bb = Bs + buf * BSZ + boff;
        const v16bf a = frag16(Ab, Ab + 8);
        v16bf bf[4];
#pragma unroll
        for (int g = 0; g < 2; ++g) {                     // groups of 4 fragments
#pragma unroll
            for (int jj = 0; jj < 4; ++jj) {              // 8x ds_load_b128 clause
                const __bf16* p = Bb + (g * 4 + jj) * 16 * LDT;
                bf[jj] = frag16(p, p + 8);
            }
#pragma unroll
            for (int jj = 0; jj < 4; ++jj)                // back-to-back WMMAs
                c[g * 4 + jj] = __builtin_amdgcn_wmma_f32_16x16x32_bf16(
                    false, a, false, bf[jj], (short)0, c[g * 4 + jj], false, false);
        }
    };

    load_regs(0);
    store_regs(0);
    __syncthreads();
    int cur = 0;
    for (int k0 = KS; k0 < DM; k0 += KS) {
        load_regs(k0);        // global -> regs (latency hidden behind WMMAs)
        compute(cur);         // LDS cur -> 8x WMMA
        store_regs(cur ^ 1);  // regs -> LDS next
        __syncthreads();      // single barrier per k-step
        cur ^= 1;
    }
    compute(cur);

#pragma unroll
    for (int j = 0; j < 8; ++j) {
        const int n = n0 + nwv * 128 + j * 16 + nm;       // C col = lane%16
        const float bias = b1[e * DH + n];
#pragma unroll
        for (int r = 0; r < 8; ++r) {                     // C row = r + (lane/16)*8
            const int m = mw * 16 + half * 8 + r;
            if (tile * TM + m < count) {
                float v = c[j][r] + bias;
                v = 0.5f * v * (1.0f + erff(v * 0.70710678118654752f));
                h[(size_t)toks[m] * DH + n] = (__bf16)v;
            }
        }
    }
}

// ------------------------------------------------------------------ FFN2 ----
// out[token][n] = h[token] @ W2[e] + b2[e] -> fp32. K=4096, N-blocks=4.
__global__ __launch_bounds__(256) void moe_ffn2_k(
    const __bf16* __restrict__ h, const float* __restrict__ W2,
    const float* __restrict__ b2, const int* __restrict__ cnt,
    const int* __restrict__ bucket, float* __restrict__ out)
{
    const int e    = blockIdx.x >> 6;
    const int tile = blockIdx.x & (MAXT - 1);
    const int count = cnt[e];
    if (tile * TM >= count) return;
    const int n0 = blockIdx.y * TNB;

    __shared__ __align__(16) __bf16 As[2 * ASZ];
    __shared__ __align__(16) __bf16 Bs[2 * BSZ];
    __shared__ int toks[TM];

    const int tid = threadIdx.x;
    if (tid < TM) {
        const int slot = tile * TM + tid;
        toks[tid] = (slot < count) ? bucket[e * NTOK + slot] : 0;
    }
    __syncthreads();

    const int am = tid >> 2, ak = (tid & 3) * 8;
    const __bf16* hrow = h + (size_t)toks[am] * DH;       // A already bf16
    const float* wcol = W2 + (size_t)e * DH * DM + n0 + tid;

    const int lane = tid & 31, wave = tid >> 5;
    const int half = lane >> 4, nm = lane & 15;
    const int mw = wave & 3, nwv = wave >> 2;
    const int aoff = (mw * 16 + nm) * LDT + half * 16;
    const int boff = (nwv * 128 + nm) * LDT + half * 16;

    v8f c[8] = {{}, {}, {}, {}, {}, {}, {}, {}};
    uint4 areg;
    float br0[16], br1[16];

    auto load_regs = [&](int k0) {
        areg = *(const uint4*)(hrow + k0 + ak);           // 8 bf16, no convert
        const float* wp = wcol + (size_t)k0 * DM;
#pragma unroll
        for (int p = 0; p < 16; ++p) {
            br0[p] = wp[(size_t)(2 * p) * DM];
            br1[p] = wp[(size_t)(2 * p + 1) * DM];
        }
    };
    auto store_regs = [&](int buf) {
        *(uint4*)&As[buf * ASZ + am * LDT + ak] = areg;   // one ds_store_b128
#pragma unroll
        for (int p = 0; p < 16; p += 2) {
            bf16x4 b4;
            b4[0] = (__bf16)br0[p];     b4[1] = (__bf16)br1[p];
            b4[2] = (__bf16)br0[p + 1]; b4[3] = (__bf16)br1[p + 1];
            *(bf16x4*)&Bs[buf * BSZ + tid * LDT + 2 * p] = b4;
        }
    };
    auto compute = [&](int buf) {
        const __bf16* Ab = As + buf * ASZ + aoff;
        const __bf16* Bb = Bs + buf * BSZ + boff;
        const v16bf a = frag16(Ab, Ab + 8);
        v16bf bf[4];
#pragma unroll
        for (int g = 0; g < 2; ++g) {
#pragma unroll
            for (int jj = 0; jj < 4; ++jj) {
                const __bf16* p = Bb + (g * 4 + jj) * 16 * LDT;
                bf[jj] = frag16(p, p + 8);
            }
#pragma unroll
            for (int jj = 0; jj < 4; ++jj)
                c[g * 4 + jj] = __builtin_amdgcn_wmma_f32_16x16x32_bf16(
                    false, a, false, bf[jj], (short)0, c[g * 4 + jj], false, false);
        }
    };

    load_regs(0);
    store_regs(0);
    __syncthreads();
    int cur = 0;
    for (int k0 = KS; k0 < DH; k0 += KS) {
        load_regs(k0);
        compute(cur);
        store_regs(cur ^ 1);
        __syncthreads();
        cur ^= 1;
    }
    compute(cur);

#pragma unroll
    for (int j = 0; j < 8; ++j) {
        const int n = n0 + nwv * 128 + j * 16 + nm;
        const float bias = b2[e * DM + n];
#pragma unroll
        for (int r = 0; r < 8; ++r) {
            const int m = mw * 16 + half * 8 + r;
            if (tile * TM + m < count)
                out[(size_t)toks[m] * DM + n] = c[j][r] + bias;
        }
    }
}

// ---------------------------------------------------------------- launch ----
extern "C" void kernel_launch(void* const* d_in, const int* in_sizes, int n_in,
                              void* d_out, int out_size, void* d_ws, size_t ws_size,
                              hipStream_t stream) {
    (void)in_sizes; (void)n_in; (void)out_size; (void)ws_size;
    const float* x  = (const float*)d_in[0];
    const float* Wg = (const float*)d_in[1];
    const float* bg = (const float*)d_in[2];
    const float* W1 = (const float*)d_in[3];
    const float* b1 = (const float*)d_in[4];
    const float* W2 = (const float*)d_in[5];
    const float* b2 = (const float*)d_in[6];
    float* out = (float*)d_out;

    // workspace: [cnt: 8 int][pad to 256B][bucket: 8*4096 int][h: 4096*4096 bf16]
    char* ws = (char*)d_ws;
    int*    cnt    = (int*)ws;
    int*    bucket = (int*)(ws + 256);
    __bf16* h      = (__bf16*)(ws + 256 + (size_t)NE * NTOK * sizeof(int));

    hipMemsetAsync(cnt, 0, NE * sizeof(int), stream);
    moe_router_k<<<NTOK / 256, 256, 0, stream>>>(x, Wg, bg, cnt, bucket);

    dim3 g1(NE * MAXT, DH / TNB);   // 512 x 16
    moe_ffn1_k<<<g1, 256, 0, stream>>>(x, W1, b1, cnt, bucket, h);

    dim3 g2(NE * MAXT, DM / TNB);   // 512 x 4
    moe_ffn2_k<<<g2, 256, 0, stream>>>(h, W2, b2, cnt, bucket, out);
}
